// Attention_15857019256917
// MI455X (gfx1250) — matrix operand order repaired and checked by reading, not compiled, not run on hardware
//
#include <hip/hip_runtime.h>
#include <hip/hip_bf16.h>

typedef _Float16 half_t;
typedef __attribute__((ext_vector_type(16))) _Float16      v16h;
typedef __attribute__((ext_vector_type(8)))  float         v8f;
typedef __attribute__((ext_vector_type(4)))  unsigned int  u32x4;
typedef __attribute__((ext_vector_type(8)))  unsigned int  u32x8;

#define DIMD 768
#define NH   12
#define HD   64
#define HH   48
#define SS   2304          // 48*48
#define MROWS 4608         // B*S

// ---------------- CDNA5 sync / async helpers ----------------

__device__ __forceinline__ unsigned lds_off_u32(const void* p) {
  // generic -> LDS address space -> 32-bit LDS byte offset
  return (unsigned)(unsigned long long)(__attribute__((address_space(3))) const void*)p;
}

__device__ __forceinline__ void wait_async0() {
#if __has_builtin(__builtin_amdgcn_s_wait_asynccnt)
  __builtin_amdgcn_s_wait_asynccnt(0);
#else
  asm volatile("s_wait_asynccnt 0x0" ::: "memory");
#endif
}

__device__ __forceinline__ void wait_tensor0() {
#if __has_builtin(__builtin_amdgcn_s_wait_tensorcnt)
  __builtin_amdgcn_s_wait_tensorcnt(0);
#else
  asm volatile("s_wait_tensorcnt 0x0" ::: "memory");
#endif
}

// per-lane async 16B global -> LDS copy (ASYNCcnt tracked)
__device__ __forceinline__ void async_cp_b128(void* lds_dst, const void* gsrc) {
  asm volatile("global_load_async_to_lds_b128 %0, %1, off"
               :: "v"(lds_off_u32(lds_dst)),
                  "v"((unsigned long long)gsrc)
               : "memory");
}

// TDM: DMA a [tile_rows x 64] f16 tile (row stride 64 elems) global -> LDS.
// D# built per ISA 8.3/8.4; groups 2/3 zero (2-D tile). TENSORcnt tracked.
__device__ __forceinline__ void tdm_load_tile_2d(unsigned lds_byte_off, const void* gaddr,
                                                 unsigned tile_rows, unsigned tensor_rows) {
  unsigned long long ga = (unsigned long long)gaddr;
  u32x4 g0;
  g0[0] = 1u;                                               // count=1 (valid user D#)
  g0[1] = lds_byte_off;                                     // lds_addr
  g0[2] = (unsigned)ga;                                     // global_addr[31:0]
  g0[3] = (unsigned)((ga >> 32) & 0x01ffffffu) | (2u << 30);// addr[56:32] | type=2
  u32x8 g1;
  g1[0] = 1u << 16;                                         // data_size=1 -> 2 bytes
  g1[1] = (64u & 0xffffu) << 16;                            // tensor_dim0[15:0]=64
  g1[2] = 0u | ((tensor_rows & 0xffffu) << 16);             // dim0 hi | tensor_dim1 lo
  g1[3] = ((tensor_rows >> 16) & 0xffffu) | (64u << 16);    // dim1 hi | tile_dim0=64
  g1[4] = tile_rows & 0xffffu;                              // tile_dim1 | tile_dim2=0
  g1[5] = 64u;                                              // tensor_dim0_stride=64
  g1[6] = 0u;
  g1[7] = 0u;
  u32x4 gz = {0u, 0u, 0u, 0u};
  asm volatile("tensor_load_to_lds %0, %1, %2, %3"
               :: "s"(g0), "s"(g1), "s"(gz), "s"(gz)
               : "memory");
}

// ---------------- WMMA fragment helpers (contiguous k order: k = 16*(lane/16) + i, wave32) ----------------

__device__ __forceinline__ v16h frag_a_load(const half_t* __restrict__ src, int ld, int lane) {
  const int m  = lane & 15;
  const int kb = (lane & 16) ? 16 : 0;
  const half_t* p = src + m * ld + kb;
  v16h a;
#pragma unroll
  for (int i = 0; i < 8; ++i) a[i] = p[i];
#pragma unroll
  for (int i = 0; i < 8; ++i) a[8 + i] = p[8 + i];
  return a;
}

__device__ __forceinline__ v16h frag_bt_load(const half_t* __restrict__ src, int ld, int lane) {
  const int n  = lane & 15;
  const int kb = (lane & 16) ? 16 : 0;
  const half_t* p = src + n * ld + kb;
  v16h b;
#pragma unroll
  for (int i = 0; i < 16; ++i) b[i] = p[i];
  return b;
}

__device__ __forceinline__ v16h frag_b_load(const half_t* __restrict__ src, int ld, int lane) {
  const int n  = lane & 15;
  const int kb = (lane & 16) ? 16 : 0;
  const half_t* p = src + kb * ld + n;
  v16h b;
#pragma unroll
  for (int i = 0; i < 16; ++i) b[i] = p[i * ld];
  return b;
}

__device__ __forceinline__ v8f wmma16(v16h a, v16h b, v8f c) {
  return __builtin_amdgcn_wmma_f32_16x16x32_f16(false, a, false, b, (short)0, c, false, false);
}

// ---------------- conversion kernels ----------------

__global__ void cvt_f16_kernel(const float* __restrict__ src, half_t* __restrict__ dst, int n) {
  int i = blockIdx.x * blockDim.x + threadIdx.x;
  if (i < n) dst[i] = (half_t)src[i];
}

__global__ void cvt_transpose_kernel(const float* __restrict__ src, half_t* __restrict__ dst,
                                     int K, int N) {
  int i = blockIdx.x * blockDim.x + threadIdx.x;
  if (i < K * N) {
    int k = i / N, n = i % N;
    dst[n * K + k] = (half_t)src[i];
  }
}

// ---------------- qkv GEMM: [4608,768] @ [768,2304] + b, scatter to q/k/v ----------------

__global__ __launch_bounds__(256) void qkv_gemm_kernel(
    const half_t* __restrict__ xh, const half_t* __restrict__ wT,
    const float* __restrict__ bias,
    half_t* __restrict__ qO, half_t* __restrict__ kO, half_t* __restrict__ vO) {
  __shared__ half_t sA[2][64 * 32];
  __shared__ half_t sB[2][128 * 32];
  const int tid  = threadIdx.x;
  const int lane = tid & 31;
  const int wave = tid >> 5;
  const int wm   = wave & 1;
  const int wn   = wave >> 1;
  const int m0   = blockIdx.y * 64;
  const int n0   = blockIdx.x * 128;

  const int arow = tid >> 2, acol = (tid & 3) * 8;
  const int brow = tid >> 1, bcol = (tid & 1) * 16;

  auto issue_tiles = [&](int buf, int kt) {
    async_cp_b128(&sA[buf][arow * 32 + acol], &xh[(long)(m0 + arow) * DIMD + kt + acol]);
    async_cp_b128(&sB[buf][brow * 32 + bcol], &wT[(long)(n0 + brow) * DIMD + kt + bcol]);
    async_cp_b128(&sB[buf][brow * 32 + bcol + 8], &wT[(long)(n0 + brow) * DIMD + kt + bcol + 8]);
  };

  v8f z = {};
  v8f acc[2][2];
#pragma unroll
  for (int i = 0; i < 2; ++i)
#pragma unroll
    for (int j = 0; j < 2; ++j) acc[i][j] = z;

  issue_tiles(0, 0);
  for (int ki = 0; ki < DIMD / 32; ++ki) {
    const int cur = ki & 1;
    wait_async0();
    __syncthreads();
    if (ki + 1 < DIMD / 32) issue_tiles(cur ^ 1, (ki + 1) * 32);

    v16h afr[2], bfr[2];
#pragma unroll
    for (int mi = 0; mi < 2; ++mi)
      afr[mi] = frag_a_load(&sA[cur][(wm * 32 + mi * 16) * 32], 32, lane);
#pragma unroll
    for (int ni = 0; ni < 2; ++ni)
      bfr[ni] = frag_bt_load(&sB[cur][(wn * 32 + ni * 16) * 32], 32, lane);
#pragma unroll
    for (int mi = 0; mi < 2; ++mi)
#pragma unroll
      for (int ni = 0; ni < 2; ++ni) acc[mi][ni] = wmma16(afr[mi], bfr[ni], acc[mi][ni]);
    __syncthreads();
  }

  // epilogue: batch / qkv-select / head are block/fragment-uniform
  const int hi  = (lane & 16) ? 8 : 0;
  const int ln  = lane & 15;
  const int b   = m0 / SS;                 // 2304 % 64 == 0 -> uniform per block
  const int sel = n0 / DIMD;               // 768 % 128 == 0 -> uniform per block
  half_t* dstA = (sel == 0) ? qO : ((sel == 1) ? kO : vO);
  const float scl = (sel == 0) ? 0.125f : 1.0f;   // fold softmax scale into q (exact)
#pragma unroll
  for (int ni = 0; ni < 2; ++ni) {
    const int nbase = n0 + wn * 32 + ni * 16;     // multiple of 16: head uniform per frag
    const int rem   = nbase % DIMD;
    const int head  = rem >> 6;
    const int d     = (rem & 63) + ln;
    const float bia = bias[nbase + ln];
    half_t* hdst = dstA + (((long)(b * NH + head)) * SS) * HD + d;
#pragma unroll
    for (int mi = 0; mi < 2; ++mi)
#pragma unroll
      for (int r = 0; r < 8; ++r) {
        int s = (m0 - b * SS) + wm * 32 + mi * 16 + r + hi;
        hdst[(long)s * HD] = (half_t)((acc[mi][ni][r] + bia) * scl);
      }
  }
}

// ---------------- decomposed rel-pos bias tables ----------------

__global__ __launch_bounds__(256) void relpos_kernel(
    const half_t* __restrict__ qA, const float* __restrict__ tabH,
    const float* __restrict__ tabW, float* __restrict__ relh, float* __restrict__ relw) {
  const int bh = blockIdx.x / HH;
  const int h  = blockIdx.x % HH;
  const half_t* qrow0 = qA + ((long)bh * SS + h * HH) * HD;
  float* outH = relh + (long)(bh * HH + h) * HH * HH;
  float* outW = relw + (long)(bh * HH + h) * HH * HH;
  for (int idx = threadIdx.x; idx < HH * HH; idx += blockDim.x) {
    int w = idx / HH, kk = idx % HH;
    const half_t* q  = qrow0 + w * HD;
    const float*  th = tabH + (h - kk + HH - 1) * HD;
    const float*  tw = tabW + (w - kk + HH - 1) * HD;
    float sh = 0.f, sw = 0.f;
#pragma unroll 8
    for (int d = 0; d < HD; ++d) {
      float qq = (float)q[d] * 8.0f;      // undo folded 1/8 (exact)
      sh += qq * th[d];
      sw += qq * tw[d];
    }
    outH[idx] = sh;
    outW[idx] = sw;
  }
}

// ---------------- flash attention: TDM double-buffered K/V, fused rel-pos bias ----------------

__global__ __launch_bounds__(128) void flash_kernel(
    const half_t* __restrict__ qA, const half_t* __restrict__ kA,
    const half_t* __restrict__ vA, const float* __restrict__ relh,
    const float* __restrict__ relw, half_t* __restrict__ aoh) {
  __shared__ half_t sK[2][32 * 64];
  __shared__ half_t sV[2][32 * 64];
  __shared__ half_t sP[4][16 * 32];
  const int tid  = threadIdx.x;
  const int lane = tid & 31;
  const int wave = tid >> 5;                       // 0..3
  const int bh   = blockIdx.y;                     // 0..23
  const int q0   = blockIdx.x * 64 + wave * 16;
  const int hi   = (lane & 16) ? 8 : 0;
  const int ln   = lane & 15;

  const half_t* qbase = qA + ((long)bh * SS + q0) * HD;
  const v16h qa0 = frag_a_load(qbase, HD, lane);
  const v16h qa1 = frag_a_load(qbase + 32, HD, lane);

  const float* rhB = relh + (long)bh * SS * HH;
  const float* rwB = relw + (long)bh * SS * HH;
  int rowbase[8];
#pragma unroll
  for (int r = 0; r < 8; ++r) rowbase[r] = (q0 + r + hi) * HH;

  float mrow[8], lrow[8];
  v8f z = {};
  v8f o[4];
#pragma unroll
  for (int f = 0; f < 4; ++f) o[f] = z;
#pragma unroll
  for (int r = 0; r < 8; ++r) { mrow[r] = -3.0e38f; lrow[r] = 0.f; }

  const half_t* kBase = kA + (long)bh * SS * HD;
  const half_t* vBase = vA + (long)bh * SS * HD;

  // prologue: wave 0 DMAs tile 0 via the Tensor Data Mover
  if (wave == 0) {
    tdm_load_tile_2d(lds_off_u32(&sK[0][0]), kBase, 32, SS);
    tdm_load_tile_2d(lds_off_u32(&sV[0][0]), vBase, 32, SS);
  }

  for (int t = 0; t < SS / 32; ++t) {
    const int cur  = t & 1;
    const int key0 = t * 32;
    wait_tensor0();              // issuing wave waits its TDMs; others pass
    __syncthreads();             // everyone sees the fresh tile
    if (wave == 0 && t + 1 < SS / 32) {   // DMA next tile into other buffer
      tdm_load_tile_2d(lds_off_u32(&sK[cur ^ 1][0]), kBase + (long)(t + 1) * 32 * HD, 32, SS);
      tdm_load_tile_2d(lds_off_u32(&sV[cur ^ 1][0]), vBase + (long)(t + 1) * 32 * HD, 32, SS);
    }
    const half_t* sKc = &sK[cur][0];
    const half_t* sVc = &sV[cur][0];

    // S = q @ k^T
    v8f s[2];
    s[0] = z; s[1] = z;
#pragma unroll
    for (int f = 0; f < 2; ++f) {
      v16h b0 = frag_bt_load(&sKc[(f * 16) * 64], 64, lane);
      v16h b1 = frag_bt_load(&sKc[(f * 16) * 64 + 32], 64, lane);
      s[f] = wmma16(qa0, b0, s[f]);
      s[f] = wmma16(qa1, b1, s[f]);
    }

    // fused decomposed rel-pos bias
    int kh[2], kw[2];
#pragma unroll
    for (int f = 0; f < 2; ++f) {
      int key = key0 + f * 16 + ln;
      kh[f] = key / HH;
      kw[f] = key % HH;
    }
#pragma unroll
    for (int f = 0; f < 2; ++f)
#pragma unroll
      for (int r = 0; r < 8; ++r)
        s[f][r] += rhB[rowbase[r] + kh[f]] + rwB[rowbase[r] + kw[f]];

    // online softmax (rows live across 16-lane groups)
    float pst0[8], pst1[8];
#pragma unroll
    for (int r = 0; r < 8; ++r) {
      float tmax = fmaxf(s[0][r], s[1][r]);
#pragma unroll
      for (int mk = 8; mk >= 1; mk >>= 1) tmax = fmaxf(tmax, __shfl_xor(tmax, mk, 32));
      float mnew = fmaxf(mrow[r], tmax);
      float corr = __expf(mrow[r] - mnew);
      float p0 = __expf(s[0][r] - mnew);
      float p1 = __expf(s[1][r] - mnew);
      float rs = p0 + p1;
#pragma unroll
      for (int mk = 8; mk >= 1; mk >>= 1) rs += __shfl_xor(rs, mk, 32);
      lrow[r] = lrow[r] * corr + rs;
      mrow[r] = mnew;
#pragma unroll
      for (int f = 0; f < 4; ++f) o[f][r] *= corr;
      pst0[r] = p0;
      pst1[r] = p1;
    }

    // C-layout -> A-layout for P via per-wave LDS staging
    half_t* myP = &sP[wave][0];
#pragma unroll
    for (int r = 0; r < 8; ++r) {
      myP[(r + hi) * 32 + ln]      = (half_t)pst0[r];
      myP[(r + hi) * 32 + 16 + ln] = (half_t)pst1[r];
    }
    v16h pa = frag_a_load(myP, 32, lane);

    // O += P @ V
#pragma unroll
    for (int f = 0; f < 4; ++f) {
      v16h bv = frag_b_load(&sVc[f * 16], 64, lane);
      o[f] = wmma16(pa, bv, o[f]);
    }
  }

  // normalize and write head-interleaved f16 output [B][S][head*64+d]
  const int b = bh / NH, head = bh % NH;
#pragma unroll
  for (int r = 0; r < 8; ++r) {
    int sq = q0 + r + hi;
    float inv = 1.0f / lrow[r];
    half_t* dst = aoh + ((long)(b * SS + sq)) * DIMD + head * HD;
#pragma unroll
    for (int f = 0; f < 4; ++f) dst[f * 16 + ln] = (half_t)(o[f][r] * inv);
  }
}

// ---------------- proj GEMM: [4608,768] @ [768,768] + b -> fp32 out ----------------

__global__ __launch_bounds__(256) void proj_gemm_kernel(
    const half_t* __restrict__ ah, const half_t* __restrict__ wT,
    const float* __restrict__ bias, float* __restrict__ out) {
  __shared__ half_t sA[2][64 * 32];
  __shared__ half_t sB[2][128 * 32];
  const int tid  = threadIdx.x;
  const int lane = tid & 31;
  const int wave = tid >> 5;
  const int wm   = wave & 1;
  const int wn   = wave >> 1;
  const int m0   = blockIdx.y * 64;
  const int n0   = blockIdx.x * 128;

  const int arow = tid >> 2, acol = (tid & 3) * 8;
  const int brow = tid >> 1, bcol = (tid & 1) * 16;

  auto issue_tiles = [&](int buf, int kt) {
    async_cp_b128(&sA[buf][arow * 32 + acol], &ah[(long)(m0 + arow) * DIMD + kt + acol]);
    async_cp_b128(&sB[buf][brow * 32 + bcol], &wT[(long)(n0 + brow) * DIMD + kt + bcol]);
    async_cp_b128(&sB[buf][brow * 32 + bcol + 8], &wT[(long)(n0 + brow) * DIMD + kt + bcol + 8]);
  };

  v8f z = {};
  v8f acc[2][2];
#pragma unroll
  for (int i = 0; i < 2; ++i)
#pragma unroll
    for (int j = 0; j < 2; ++j) acc[i][j] = z;

  issue_tiles(0, 0);
  for (int ki = 0; ki < DIMD / 32; ++ki) {
    const int cur = ki & 1;
    wait_async0();
    __syncthreads();
    if (ki + 1 < DIMD / 32) issue_tiles(cur ^ 1, (ki + 1) * 32);

    v16h afr[2], bfr[2];
#pragma unroll
    for (int mi = 0; mi < 2; ++mi)
      afr[mi] = frag_a_load(&sA[cur][(wm * 32 + mi * 16) * 32], 32, lane);
#pragma unroll
    for (int ni = 0; ni < 2; ++ni)
      bfr[ni] = frag_bt_load(&sB[cur][(wn * 32 + ni * 16) * 32], 32, lane);
#pragma unroll
    for (int mi = 0; mi < 2; ++mi)
#pragma unroll
      for (int ni = 0; ni < 2; ++ni) acc[mi][ni] = wmma16(afr[mi], bfr[ni], acc[mi][ni]);
    __syncthreads();
  }

  const int hi = (lane & 16) ? 8 : 0;
  const int ln = lane & 15;
#pragma unroll
  for (int mi = 0; mi < 2; ++mi)
#pragma unroll
    for (int ni = 0; ni < 2; ++ni)
#pragma unroll
      for (int r = 0; r < 8; ++r) {
        int m = m0 + wm * 32 + mi * 16 + r + hi;
        int n = n0 + wn * 32 + ni * 16 + ln;
        out[(long)m * DIMD + n] = acc[mi][ni][r] + bias[n];
      }
}

// ---------------- host-side launcher ----------------

extern "C" void kernel_launch(void* const* d_in, const int* in_sizes, int n_in,
                              void* d_out, int out_size, void* d_ws, size_t ws_size,
                              hipStream_t stream) {
  const float* x      = (const float*)d_in[0];
  const float* qkv_w  = (const float*)d_in[1];
  const float* qkv_b  = (const float*)d_in[2];
  const float* proj_w = (const float*)d_in[3];
  const float* proj_b = (const float*)d_in[4];
  const float* rph    = (const float*)d_in[5];
  const float* rpw    = (const float*)d_in[6];
  float* out = (float*)d_out;

  char* ws = (char*)d_ws;
  half_t* xh     = (half_t*)(ws + 0);           // 4608*768 f16
  half_t* qkvwT  = (half_t*)(ws + 7077888);     // 2304*768 f16 (transposed)
  half_t* projwT = (half_t*)(ws + 10616832);    // 768*768 f16 (transposed)
  half_t* qS     = (half_t*)(ws + 11796480);    // 24*2304*64 f16 (scaled)
  half_t* kS     = (half_t*)(ws + 18874368);
  half_t* vS     = (half_t*)(ws + 25952256);
  half_t* aoh    = (half_t*)(ws + 33030144);    // [B,S,768] head-interleaved f16
  float*  relh   = (float*)(ws + 40108032);     // 24*2304*48 fp32
  float*  relw   = (float*)(ws + 50724864);

  cvt_f16_kernel<<<(MROWS * DIMD + 255) / 256, 256, 0, stream>>>(x, xh, MROWS * DIMD);
  cvt_transpose_kernel<<<(768 * 2304 + 255) / 256, 256, 0, stream>>>(qkv_w, qkvwT, 768, 2304);
  cvt_transpose_kernel<<<(768 * 768 + 255) / 256, 256, 0, stream>>>(proj_w, projwT, 768, 768);
  qkv_gemm_kernel<<<dim3(2304 / 128, MROWS / 64), 256, 0, stream>>>(xh, qkvwT, qkv_b, qS, kS, vS);
  relpos_kernel<<<24 * HH, 256, 0, stream>>>(qS, rph, rpw, relh, relw);
  flash_kernel<<<dim3(SS / 64, 24), 128, 0, stream>>>(qS, kS, vS, relh, relw, aoh);
  proj_gemm_kernel<<<dim3(DIMD / 128, MROWS / 64), 256, 0, stream>>>(aoh, projwT, proj_b, out);
}
